// EngramLayer_15109694947887
// MI455X (gfx1250) — matrix operand order repaired and checked by reading, not compiled, not run on hardware
//
#include <hip/hip_runtime.h>

// ---------------- problem constants ----------------
#define B_   4
#define S_   2048
#define M_   4
#define D_   2048
#define E_   1024
#define H_   8
#define DPH_ 128
#define R_   (B_ * S_)        // 8192 rows (b,s)
#define NTOT_ (D_ + M_ * D_)  // 10240 weight rows: [value | keys(m=0..3)]
#define LDSROW_ 1032          // padded LDS row (516 dwords -> 4-bank skew, conflict-free)

typedef __bf16 bf16;
typedef __attribute__((ext_vector_type(16))) __bf16 v16bf;
typedef __attribute__((ext_vector_type(8)))  __bf16 v8bf;
typedef __attribute__((ext_vector_type(8)))  float  v8f;

// prime-table offsets (cumsum of PRIMES[:-1])
__device__ const int kHashOff[8] = {0, 130003, 260024, 390051, 520094, 650145, 780202, 910271};

// ---------------- kernel 0: weights f32 -> bf16, concatenated [NTOT, E] ----------------
__global__ __launch_bounds__(256)
void engram_wcvt(const float* __restrict__ w_v, const float* __restrict__ w_k,
                 bf16* __restrict__ Wbf) {
  size_t i = (size_t)blockIdx.x * 256 + threadIdx.x;      // [n][e] row-major
  if (i >= (size_t)NTOT_ * E_) return;
  size_t n = i >> 10;
  float f = (n < D_) ? w_v[i] : w_k[i - (size_t)D_ * E_];
  Wbf[i] = (bf16)f;
}

// ---------------- kernel 1: hashed gather + f32 -> bf16, emb [R, E] ----------------
__global__ __launch_bounds__(256)
void engram_gather(const int* __restrict__ hash_idx, const float* __restrict__ table,
                   bf16* __restrict__ emb) {
  size_t i = (size_t)blockIdx.x * 256 + threadIdx.x;      // [r][e]
  if (i >= (size_t)R_ * E_) return;
  int e = (int)(i & (E_ - 1));
  size_t r = i >> 10;
  int h = e >> 7;
  int dph = e & (DPH_ - 1);
  int idx = hash_idx[r * H_ + h] + kHashOff[h];
  emb[i] = (bf16)table[(size_t)idx * DPH_ + dph];
}

// ---------------- kernel 2: fused WMMA GEMM + gating epilogue ----------------
// grid (5, R/16): blockIdx.x==0 -> value GEMM (writes value[R,D])
//                 blockIdx.x==q>=1 -> keys GEMM for branch m=q-1, fully fused
//                 RMSNorm/score/gate epilogue (writes only gate[R,M]; keys never hit HBM).
// block = 512 threads = 16 waves; wave tile 16 rows x 128 cols; A tile staged in LDS.
__global__ __launch_bounds__(512)
void engram_gemm_fused(const bf16* __restrict__ A, const bf16* __restrict__ W,
                       const float* __restrict__ hidden,
                       const float* __restrict__ norm_h_w, const float* __restrict__ norm_k_w,
                       float* __restrict__ value, float* __restrict__ gate_out) {
  __shared__ __align__(16) unsigned char smem_raw[16 * LDSROW_ * 2];
  bf16* Atile = (bf16*)smem_raw;

  const int q     = blockIdx.x;              // 0 = value, 1..4 = keys branch q-1
  const int row0  = blockIdx.y << 4;
  const int lane  = threadIdx.x & 31;
  const int wave  = threadIdx.x >> 5;
  const int ncol0 = wave << 7;               // 128 cols per wave

  // cooperative A-tile fill: 16 rows x 1024 bf16 (contiguous 32KB in global)
  for (int c = threadIdx.x; c < 2048; c += 512) {
    int row = c >> 7;
    int off = (c & 127) << 3;                // 8 bf16 = 16B chunks
    *(uint4*)(Atile + row * LDSROW_ + off) =
        *(const uint4*)(A + (size_t)(row0 + row) * E_ + off);
  }
  __syncthreads();

  // A fragment (16-bit A 16x32): lane<16 -> K 0..15; lane>=16 -> K 16..31
  const int akh  = (lane >> 4) << 4;
  // B fragment (16-bit B 32x16): lane<16 -> col=lane, K 0..15; lane>=16 -> col=lane-16, K 16..31
  const int bcol = lane & 15;
  const int bkh  = (lane >> 4) << 4;

  const bf16* Arow = Atile + (lane & 15) * LDSROW_;
  const bf16* Wq   = W + (size_t)q * D_ * E_;

  v8f acc[8];
#pragma unroll
  for (int j = 0; j < 8; ++j) acc[j] = v8f{0};

  for (int k0 = 0; k0 < E_; k0 += 32) {
    v8bf alo = *(const v8bf*)(Arow + k0 + akh);
    v8bf ahi = *(const v8bf*)(Arow + k0 + 8 + akh);
    v16bf a  = __builtin_shufflevector(alo, ahi,
                 0,1,2,3,4,5,6,7,8,9,10,11,12,13,14,15);
#pragma unroll
    for (int j = 0; j < 8; ++j) {
      const bf16* Wp = Wq + (size_t)(ncol0 + (j << 4) + bcol) * E_ + k0 + bkh;
      v16bf b = *(const v16bf*)Wp;
      acc[j] = __builtin_amdgcn_wmma_f32_16x16x32_bf16(
                 false, a, false, b, (short)0, acc[j], false, false);
    }
  }

  // C layout: VGPR v -> row v (lanes 0-15) / row v+8 (lanes 16-31); col = lane & 15
  const int hi8 = (lane >> 4) << 3;
  const int cl  = lane & 15;

  if (q == 0) {
    // ---- value branch: store C ----
#pragma unroll
    for (int j = 0; j < 8; ++j) {
      int col = ncol0 + (j << 4) + cl;
#pragma unroll
      for (int v = 0; v < 8; ++v)
        value[(size_t)(row0 + hi8 + v) * D_ + col] = acc[j][v];
    }
  } else {
    // ---- keys branch: fused RMSNorm(keys), RMSNorm(hidden), score, gate ----
    const int m = q - 1;
    float psk[8], psh[8], psc[8];
#pragma unroll
    for (int v = 0; v < 8; ++v) { psk[v] = 0.f; psh[v] = 0.f; psc[v] = 0.f; }

#pragma unroll
    for (int j = 0; j < 8; ++j) {
      int col = ncol0 + (j << 4) + cl;
      float wkn = norm_k_w[m * D_ + col];
      float whn = norm_h_w[m * D_ + col];
#pragma unroll
      for (int v = 0; v < 8; ++v) {
        float kv = acc[j][v];
        float hv = hidden[((size_t)(row0 + hi8 + v) * M_ + m) * D_ + col];
        psk[v] = fmaf(kv, kv, psk[v]);
        psh[v] = fmaf(hv, hv, psh[v]);
        psc[v] = fmaf(kv * wkn, hv * whn, psc[v]);
      }
    }
    // reduce across the 16 lanes holding the same rows (xor stays within half)
#pragma unroll
    for (int off = 1; off < 16; off <<= 1) {
#pragma unroll
      for (int v = 0; v < 8; ++v) {
        psk[v] += __shfl_xor(psk[v], off, 32);
        psh[v] += __shfl_xor(psh[v], off, 32);
        psc[v] += __shfl_xor(psc[v], off, 32);
      }
    }
    __syncthreads();                 // all waves done reading Atile; reuse LDS
    float* sK = (float*)smem_raw;           // [16 waves][16 rows]
    float* sH = sK + 256;
    float* sC = sH + 256;
    if (cl == 0) {
#pragma unroll
      for (int v = 0; v < 8; ++v) {
        sK[wave * 16 + hi8 + v] = psk[v];
        sH[wave * 16 + hi8 + v] = psh[v];
        sC[wave * 16 + hi8 + v] = psc[v];
      }
    }
    __syncthreads();
    if (threadIdx.x < 16) {
      int row = threadIdx.x;
      float sk = 0.f, sh = 0.f, sc = 0.f;
#pragma unroll
      for (int w = 0; w < 16; ++w) {
        sk += sK[w * 16 + row]; sh += sH[w * 16 + row]; sc += sC[w * 16 + row];
      }
      const float invD = 1.0f / (float)D_;
      float ik = rsqrtf(sk * invD + 1.1920929e-07f);
      float ih = rsqrtf(sh * invD + 1.1920929e-07f);
      float score = sc * ik * ih * 0.022097086912079608f;   // 1/sqrt(2048)
      float sgn = (score > 0.f) ? 1.f : ((score < 0.f) ? -1.f : 0.f);
      float g = sqrtf(fmaxf(fabsf(score), 1e-6f)) * sgn;
      gate_out[(row0 + row) * M_ + m] = 1.f / (1.f + expf(-g));
    }
  }
}

// ---------------- block reduction helper (wave32, 256-thread blocks) ----------------
__device__ __forceinline__ float block_sum(float v, float* s) {
#pragma unroll
  for (int off = 16; off; off >>= 1) v += __shfl_xor(v, off, 32);
  const int lane = threadIdx.x & 31, wv = threadIdx.x >> 5;
  if (lane == 0) s[wv] = v;
  __syncthreads();
  float r = s[lane & 7];
#pragma unroll
  for (int off = 4; off; off >>= 1) r += __shfl_xor(r, off, 32);
  __syncthreads();
  return r;
}

// ---------------- kernel 3: apply gate -> gated (d_out), inv_rms for conv norm ----------------
// sum(value^2) is m-invariant, computed once per (b,s) row.
__global__ __launch_bounds__(256)
void engram_apply(const float* __restrict__ value, const float* __restrict__ gate,
                  float* __restrict__ out, float* __restrict__ inv_rms) {
  __shared__ float sred[8];
  const int r = blockIdx.x;
  float vv[8];
  float sv = 0.f;
#pragma unroll
  for (int i = 0; i < 8; ++i) {
    int d = threadIdx.x + (i << 8);
    vv[i] = value[(size_t)r * D_ + d];
    sv = fmaf(vv[i], vv[i], sv);
  }
  sv = block_sum(sv, sred);
  const float svD = sv * (1.0f / (float)D_);
#pragma unroll
  for (int m = 0; m < M_; ++m) {
    float g = gate[r * M_ + m];
    size_t base = ((size_t)r * M_ + m) * D_;
#pragma unroll
    for (int i = 0; i < 8; ++i) {
      int d = threadIdx.x + (i << 8);
      out[base + d] = g * vv[i];
    }
    if (threadIdx.x == 0)
      inv_rms[r * M_ + m] = rsqrtf(g * g * svD + 1e-5f);
  }
}

// ---------------- kernel 4: dilated causal depthwise conv + SiLU + residual (in place) ----------------
// one thread owns one (b,m,d) channel over all S timesteps -> no cross-block hazard on d_out.
__global__ __launch_bounds__(256)
void engram_conv(const float* __restrict__ hidden, const float* __restrict__ conv_norm_w,
                 const float* __restrict__ conv_w, const float* __restrict__ inv_rms,
                 float* __restrict__ io) {
  const int dchunk = blockIdx.x & 7;
  const int m = (blockIdx.x >> 3) & (M_ - 1);
  const int b = blockIdx.x >> 5;
  const int d = (dchunk << 8) + threadIdx.x;
  const int c = m * D_ + d;

  const float w0 = conv_w[c * 4 + 0], w1 = conv_w[c * 4 + 1];
  const float w2 = conv_w[c * 4 + 2], w3 = conv_w[c * 4 + 3];
  const float cw = conv_norm_w[c];

  float hist[9];
#pragma unroll
  for (int i = 0; i < 9; ++i) hist[i] = 0.f;

  const size_t base = ((size_t)b * S_) * (M_ * D_) + (size_t)m * D_ + d;
  for (int t = 0; t < S_; ++t) {
    size_t idx = base + (size_t)t * (M_ * D_);
    float g  = io[idx];                                    // gated
    float xn = g * inv_rms[(b * S_ + t) * M_ + m] * cw;    // RMSNorm(gated)
    // conv[t] = w3*x[t] + w2*x[t-3] + w1*x[t-6] + w0*x[t-9]
    float cv = fmaf(w3, xn, fmaf(w2, hist[2], fmaf(w1, hist[5], w0 * hist[8])));
    float sil = cv / (1.f + expf(-cv));                    // SiLU
    io[idx] = hidden[idx] + g + sil;
#pragma unroll
    for (int i = 8; i > 0; --i) hist[i] = hist[i - 1];
    hist[0] = xn;
  }
}

// ---------------- host launcher ----------------
extern "C" void kernel_launch(void* const* d_in, const int* in_sizes, int n_in,
                              void* d_out, int out_size, void* d_ws, size_t ws_size,
                              hipStream_t stream) {
  (void)in_sizes; (void)n_in; (void)out_size; (void)ws_size;
  const int*   hash_idx    = (const int*)  d_in[0];
  const float* hidden      = (const float*)d_in[1];
  const float* emb_table   = (const float*)d_in[2];
  const float* w_v         = (const float*)d_in[3];
  const float* w_k         = (const float*)d_in[4];
  const float* norm_h_w    = (const float*)d_in[5];
  const float* norm_k_w    = (const float*)d_in[6];
  const float* conv_norm_w = (const float*)d_in[7];
  const float* conv_w      = (const float*)d_in[8];
  float* out = (float*)d_out;

  // workspace layout (bytes); total ~100 MB
  char* ws = (char*)d_ws;
  const size_t off_w    = 0;                                   // 10240*1024 bf16 = 20.97 MB
  const size_t off_emb  = off_w    + (size_t)NTOT_ * E_ * 2;   //  8192*1024 bf16 = 16.78 MB
  const size_t off_val  = off_emb  + (size_t)R_ * E_ * 2;      //  8192*2048 f32  = 67.1  MB
  const size_t off_gate = off_val  + (size_t)R_ * D_ * 4;      //  8192*4    f32
  const size_t off_irms = off_gate + (size_t)R_ * M_ * 4;      //  8192*4    f32
  bf16*  Wbf     = (bf16*)(ws + off_w);
  bf16*  emb     = (bf16*)(ws + off_emb);
  float* value   = (float*)(ws + off_val);
  float* gate    = (float*)(ws + off_gate);
  float* inv_rms = (float*)(ws + off_irms);

  // 0) weight convert
  {
    size_t n = (size_t)NTOT_ * E_;
    engram_wcvt<<<dim3((unsigned)((n + 255) / 256)), dim3(256), 0, stream>>>(w_v, w_k, Wbf);
  }
  // 1) hashed gather + convert
  {
    size_t n = (size_t)R_ * E_;
    engram_gather<<<dim3((unsigned)((n + 255) / 256)), dim3(256), 0, stream>>>(hash_idx, emb_table, emb);
  }
  // 2) fused value-GEMM + keys-GEMM/gating (keys never materialized to HBM)
  engram_gemm_fused<<<dim3(1 + M_, R_ / 16), dim3(512), 0, stream>>>(
      emb, Wbf, hidden, norm_h_w, norm_k_w, value, gate);
  // 3) gated = gate*value -> d_out; inv_rms for conv
  engram_apply<<<dim3(R_), dim3(256), 0, stream>>>(value, gate, out, inv_rms);
  // 4) in-place conv + SiLU + residual
  engram_conv<<<dim3(B_ * M_ * (D_ / 256)), dim3(256), 0, stream>>>(
      hidden, conv_norm_w, conv_w, inv_rms, out);
}
